// NetworkForAGraphWithAttributesPeriodic_6571299963039
// MI455X (gfx1250) — matrix operand order repaired and checked by reading, not compiled, not run on hardware
//
#include <hip/hip_runtime.h>

typedef __attribute__((ext_vector_type(16))) _Float16 v16h;
typedef __attribute__((ext_vector_type(8)))  float    v8f;
typedef __attribute__((ext_vector_type(2)))  float    v2f;

#define N_NODES 16000
#define N_EDGES 512000
#define CCH 32
#define SHD 9
#define NBASIS 10
#define LAYERS 3
#define FEAT (CCH*SHD)   /* 288 */

__device__ __forceinline__ float silu_f(float x){ return x / (1.f + __expf(-x)); }
__device__ __forceinline__ float sigm_f(float x){ return 1.f / (1.f + __expf(-x)); }

// Plain fire-and-forget FP32 atomic add (GLOBAL_ATOMIC_ADD_F32, no return, STOREcnt).
__device__ __forceinline__ void atomic_add_f32(float* p, float v) {
  asm volatile("global_atomic_add_f32 %0, %1, off" :: "v"(p), "v"(v) : "memory");
}

// ---------------------------------------------------------------------------
// Weight prep: pad + transpose to column-major f16 panels (lane = column,
// K contiguous) so B fragments are two contiguous 16B loads per lane.
//   W1p[l][n<112][k<32], W2p[l][n<32][k<128], Wattrp[l][n<32][k<32]
// ---------------------------------------------------------------------------
__global__ void prep_weights_kernel(const float* __restrict__ W1,
                                    const float* __restrict__ W2,
                                    const float* __restrict__ Wattr,
                                    _Float16* __restrict__ W1p,
                                    _Float16* __restrict__ W2p,
                                    _Float16* __restrict__ Wattrp) {
  int t = blockIdx.x * blockDim.x + threadIdx.x;
  const int n1 = LAYERS * 112 * 32;
  const int n2 = LAYERS * 32 * 128;
  const int n3 = LAYERS * 32 * 32;
  if (t < n1) {
    int l = t / (112*32); int rem = t % (112*32); int n = rem / 32; int k = rem % 32;
    float v = (k < NBASIS && n < 100) ? W1[(l*NBASIS + k)*100 + n] : 0.f;
    W1p[t] = (_Float16)v;
  } else if (t < n1 + n2) {
    int u = t - n1;
    int l = u / (32*128); int rem = u % (32*128); int n = rem / 128; int k = rem % 128;
    float v = (k < 100) ? W2[(l*100 + k)*32 + n] : 0.f;
    W2p[u] = (_Float16)v;
  } else if (t < n1 + n2 + n3) {
    int u = t - n1 - n2;
    int l = u / (32*32); int rem = u % (32*32); int n = rem / 32; int k = rem % 32;
    float v = (k < 13) ? Wattr[(l*13 + k)*32 + n] : 0.f;
    Wattrp[u] = (_Float16)v;
  }
}

// ---------------------------------------------------------------------------
// Per-edge geometry: spherical harmonics l=0,1,2 (f32, reused in messages),
// edge attr concat -> f16 (K padded 13->16), radial basis -> f16 (10->16).
// ---------------------------------------------------------------------------
__global__ void edge_features_kernel(const float* __restrict__ edge_vec,
                                     const float* __restrict__ edge_attr,
                                     float* __restrict__ sh,
                                     _Float16* __restrict__ emb16,
                                     _Float16* __restrict__ ea16) {
  int e = blockIdx.x * blockDim.x + threadIdx.x;
  if (e >= N_EDGES) return;
  float vx = edge_vec[e*3+0], vy = edge_vec[e*3+1], vz = edge_vec[e*3+2];
  float r2 = vx*vx + vy*vy + vz*vz + 1e-12f;
  float d  = sqrtf(r2);
  float inv = 1.f / d;
  float x = vx*inv, y = vy*inv, z = vz*inv;
  const float s3    = 1.7320508075688772f;   // sqrt(3)
  const float s15   = 3.8729833462074170f;   // sqrt(15)
  const float s5_4  = 1.1180339887498949f;   // sqrt(5)/2
  const float s15_4 = 1.9364916731037085f;   // sqrt(15)/2
  float shl[9];
  shl[0] = 1.f;        shl[1] = s3*x;        shl[2] = s3*y;  shl[3] = s3*z;
  shl[4] = s15*x*y;    shl[5] = s15*y*z;     shl[6] = s5_4*(3.f*z*z - 1.f);
  shl[7] = s15*z*x;    shl[8] = s15_4*(x*x - y*y);
  #pragma unroll
  for (int k = 0; k < 9; ++k) sh[e*9 + k] = shl[k];

  // radial basis: centers 5*(i+1)/11, half-cosine window, * sqrt(10)
  const float step = 5.0f / 11.0f;
  const float sq10 = 3.1622776601683795f;
  const float halfpi = 1.5707963267948966f;
  #pragma unroll
  for (int i = 0; i < NBASIS; ++i) {
    float diff = (d - step*(float)(i+1)) / step;
    float win = (diff > -1.f && diff < 1.f) ? 1.f : 0.f;
    emb16[e*16 + i] = (_Float16)(__cosf(halfpi*diff) * win * sq10);
  }
  #pragma unroll
  for (int i = NBASIS; i < 16; ++i) emb16[e*16 + i] = (_Float16)0.f;

  #pragma unroll
  for (int i = 0; i < 4; ++i) ea16[e*16 + i] = (_Float16)edge_attr[e*4 + i];
  #pragma unroll
  for (int k = 0; k < 9; ++k) ea16[e*16 + 4 + k] = (_Float16)shl[k];
  #pragma unroll
  for (int i = 13; i < 16; ++i) ea16[e*16 + i] = (_Float16)0.f;
}

// ---------------------------------------------------------------------------
// Edge MLP via WMMA f16: w = (silu(emb@W1+b1) @ W2) * (ea @ Wattr) / sqrt(32)
// One wave per 16-edge tile; 8 waves / 256-thread block.
// A f16 16x32 layout: lanes 0-15 row M, K {0..7,16..23}; lanes 16-31 K {8..15,24..31}.
// B layout: lane = column N (mod 16), K {0..7,16..23} (lanes 0-15) / {8..15,24..31} (lanes 16-31), as A.
// C/D f32: lane = column N, VGPR r = row hi*8+r.
// ---------------------------------------------------------------------------
#define EW_WAVES 8
__global__ __launch_bounds__(256) void edge_weight_kernel(
    const _Float16* __restrict__ emb16, const _Float16* __restrict__ ea16,
    const _Float16* __restrict__ W1p,   const _Float16* __restrict__ W2p,
    const _Float16* __restrict__ Wattrp,const float* __restrict__ b1,
    float* __restrict__ wout) {
  __shared__ _Float16 h_lds[EW_WAVES][16][128];
  const int wave = threadIdx.x >> 5;
  const int lane = threadIdx.x & 31;
  const int row  = lane & 15;
  const int hi   = lane >> 4;
  const int e0   = (blockIdx.x * EW_WAVES + wave) * 16;

  // A fragment: emb tile (K real = 10, padded; K>=16 halves are zero)
  v16h a_emb;
  {
    const _Float16* p = emb16 + (e0 + row)*16 + hi*8;
    #pragma unroll
    for (int i = 0; i < 8; ++i) a_emb[i] = p[i];
    #pragma unroll
    for (int i = 8; i < 16; ++i) a_emb[i] = (_Float16)0.f;
  }

  // h = silu(emb @ W1 + b1): 7 output tiles of 16 cols (100 padded to 112)
  #pragma unroll
  for (int nt = 0; nt < 7; ++nt) {
    v16h b;
    const _Float16* wp = W1p + (nt*16 + row)*32 + hi*8;
    #pragma unroll
    for (int i = 0; i < 16; ++i) b[i] = wp[i < 8 ? i : i + 8];
    v8f c = {};
    c = __builtin_amdgcn_wmma_f32_16x16x32_f16(false, a_emb, false, b,
                                               (short)0, c, false, false);
    const int colg = nt*16 + row;
    const float bias = (colg < 100) ? b1[colg] : 0.f;
    #pragma unroll
    for (int r = 0; r < 8; ++r)
      h_lds[wave][hi*8 + r][colg] = (_Float16)silu_f(c[r] + bias);
  }
  // zero K padding cols 112..127
  #pragma unroll
  for (int r = 0; r < 8; ++r) h_lds[wave][hi*8 + r][112 + row] = (_Float16)0.f;
  __syncthreads();

  // w_pre = h @ W2 : K=128 in 4 chunks of 32, 2 output tiles (32 channels)
  v8f acc[2] = {};
  #pragma unroll
  for (int kc = 0; kc < 4; ++kc) {
    v16h a;
    #pragma unroll
    for (int i = 0; i < 8; ++i) a[i]     = h_lds[wave][row][kc*32 + hi*8 + i];
    #pragma unroll
    for (int i = 0; i < 8; ++i) a[8 + i] = h_lds[wave][row][kc*32 + 16 + hi*8 + i];
    #pragma unroll
    for (int nt = 0; nt < 2; ++nt) {
      v16h b;
      const _Float16* wp = W2p + (nt*16 + row)*128 + kc*32 + hi*8;
      #pragma unroll
      for (int i = 0; i < 16; ++i) b[i] = wp[i < 8 ? i : i + 8];
      acc[nt] = __builtin_amdgcn_wmma_f32_16x16x32_f16(false, a, false, b,
                                                       (short)0, acc[nt], false, false);
    }
  }

  // attr = ea @ Wattr
  v8f attr[2];
  {
    v16h a;
    const _Float16* p = ea16 + (e0 + row)*16 + hi*8;
    #pragma unroll
    for (int i = 0; i < 8; ++i) a[i] = p[i];
    #pragma unroll
    for (int i = 8; i < 16; ++i) a[i] = (_Float16)0.f;
    #pragma unroll
    for (int nt = 0; nt < 2; ++nt) {
      v16h b;
      const _Float16* wp = Wattrp + (nt*16 + row)*32 + hi*8;
      #pragma unroll
      for (int i = 0; i < 16; ++i) b[i] = wp[i < 8 ? i : i + 8];
      v8f c = {};
      attr[nt] = __builtin_amdgcn_wmma_f32_16x16x32_f16(false, a, false, b,
                                                        (short)0, c, false, false);
    }
  }

  // w = w_pre * attr, fold 1/sqrt(NUM_NEIGHBORS)
  const float sc = 0.17677669529663687f;
  #pragma unroll
  for (int nt = 0; nt < 2; ++nt) {
    const int c = nt*16 + row;
    #pragma unroll
    for (int r = 0; r < 8; ++r) {
      const int e = e0 + hi*8 + r;
      wout[e*32 + c] = acc[nt][r] * attr[nt][r] * sc;
    }
  }
}

// ---------------------------------------------------------------------------
// msg[e,c,:] = x[src,c,:] * sh[e,:] * w[e,c]; scatter-add into agg[dst,c,:].
// Lane = channel, wave = edge: w/x loads coalesced, sh loads wave-uniform.
// ---------------------------------------------------------------------------
__global__ __launch_bounds__(256) void message_scatter_kernel(
    const float* __restrict__ x, const float* __restrict__ sh,
    const float* __restrict__ w, const int* __restrict__ edge_src,
    const int* __restrict__ edge_dst, float* __restrict__ agg) {
  long t = (long)blockIdx.x * blockDim.x + threadIdx.x;
  int e = (int)(t >> 5);
  int c = (int)(t & 31);
  if (e >= N_EDGES) return;
  int src = edge_src[e];
  int dst = edge_dst[e];
  float wv = w[e*32 + c];
  const float* xp = x + (long)src*FEAT + c*SHD;
  float*       ap = agg + (long)dst*FEAT + c*SHD;
  const float* shp = sh + (long)e*SHD;
  #pragma unroll
  for (int k = 0; k < SHD; ++k)
    atomic_add_f32(ap + k, xp[k] * shp[k] * wv);
}

// ---------------------------------------------------------------------------
// Node update via f32 WMMA (16x16x4, 8 K-steps): rows = (node,k) pairs (M),
// K = input channel c, N = output channel d.
//   xn = (x*node_attr) @ Wself + agg @ Wout     (both chained into same acc)
// ---------------------------------------------------------------------------
__global__ __launch_bounds__(256) void node_update_kernel(
    const float* __restrict__ x, const float* __restrict__ agg,
    const float* __restrict__ node_attr,
    const float* __restrict__ Wself, const float* __restrict__ Wout,
    float* __restrict__ xn) {
  const int wave = threadIdx.x >> 5;
  const int lane = threadIdx.x & 31;
  const int row  = lane & 15;
  const int hi   = lane >> 4;
  const int m0   = (blockIdx.x * 8 + wave) * 16;   // 144000 rows total, divides exactly
  const int M    = m0 + row;
  const int n_a  = M / SHD;
  const int k_a  = M % SHD;
  const float na = node_attr[n_a];

  v8f acc[2] = {};
  #pragma unroll
  for (int kc = 0; kc < 8; ++kc) {           // x @ Wself (scaled by node_attr)
    const int c0 = kc*4 + hi*2;
    v2f a;
    a[0] = x[(long)n_a*FEAT + (c0+0)*SHD + k_a] * na;
    a[1] = x[(long)n_a*FEAT + (c0+1)*SHD + k_a] * na;
    #pragma unroll
    for (int nt = 0; nt < 2; ++nt) {
      v2f b;
      b[0] = Wself[(c0+0)*32 + nt*16 + row];
      b[1] = Wself[(c0+1)*32 + nt*16 + row];
      acc[nt] = __builtin_amdgcn_wmma_f32_16x16x4_f32(false, a, false, b,
                                                      (short)0, acc[nt], false, false);
    }
  }
  #pragma unroll
  for (int kc = 0; kc < 8; ++kc) {           // agg @ Wout
    const int c0 = kc*4 + hi*2;
    v2f a;
    a[0] = agg[(long)n_a*FEAT + (c0+0)*SHD + k_a];
    a[1] = agg[(long)n_a*FEAT + (c0+1)*SHD + k_a];
    #pragma unroll
    for (int nt = 0; nt < 2; ++nt) {
      v2f b;
      b[0] = Wout[(c0+0)*32 + nt*16 + row];
      b[1] = Wout[(c0+1)*32 + nt*16 + row];
      acc[nt] = __builtin_amdgcn_wmma_f32_16x16x4_f32(false, a, false, b,
                                                      (short)0, acc[nt], false, false);
    }
  }
  #pragma unroll
  for (int nt = 0; nt < 2; ++nt) {
    const int d = nt*16 + row;
    #pragma unroll
    for (int r = 0; r < 8; ++r) {
      const int Mr = m0 + hi*8 + r;
      xn[(long)(Mr/SHD)*FEAT + d*SHD + (Mr % SHD)] = acc[nt][r];
    }
  }
}

// Gated nonlinearity: s = x[:, :, 0]; x = [silu(s), x[:,:,1:]*sigmoid(s)]
__global__ void gate_kernel(float* __restrict__ x) {
  int t = blockIdx.x * blockDim.x + threadIdx.x;
  if (t >= N_NODES * CCH) return;
  float* p = x + (long)t * SHD;
  float s = p[0];
  float sg = sigm_f(s);
  p[0] = s * sg;
  #pragma unroll
  for (int k = 1; k < SHD; ++k) p[k] *= sg;
}

__global__ void zero_kernel(float* __restrict__ p, long n) {
  long t = (long)blockIdx.x * blockDim.x + threadIdx.x;
  if (t < n) p[t] = 0.f;
}

// Final per-graph segment sum, / sqrt(NUM_NODES)
__global__ void output_kernel(const float* __restrict__ x,
                              const int* __restrict__ batch,
                              float* __restrict__ out) {
  long t = (long)blockIdx.x * blockDim.x + threadIdx.x;
  if (t >= (long)N_NODES * FEAT) return;
  int n = (int)(t / FEAT);
  int j = (int)(t % FEAT);
  atomic_add_f32(out + batch[n]*FEAT + j, x[t] * 0.05f);
}

extern "C" void kernel_launch(void* const* d_in, const int* in_sizes, int n_in,
                              void* d_out, int out_size, void* d_ws, size_t ws_size,
                              hipStream_t stream) {
  const float* node_input = (const float*)d_in[0];
  const float* node_attr  = (const float*)d_in[1];
  const float* edge_vec   = (const float*)d_in[2];
  const float* edge_attr  = (const float*)d_in[3];
  const float* W1         = (const float*)d_in[4];
  const float* b1         = (const float*)d_in[5];
  const float* W2         = (const float*)d_in[6];
  const float* Wattr      = (const float*)d_in[7];
  const float* Wself      = (const float*)d_in[8];
  const float* Wout       = (const float*)d_in[9];
  const int*   edge_index = (const int*)d_in[10];
  const int*   batch      = (const int*)d_in[11];
  const int* edge_dst = edge_index;             // edge_index[0]
  const int* edge_src = edge_index + N_EDGES;   // edge_index[1]
  float* out = (float*)d_out;

  char* ws = (char*)d_ws;
  size_t off = 0;
  auto alloc = [&](size_t bytes) -> void* {
    void* p = ws + off;
    off = (off + bytes + 255) & ~(size_t)255;
    return p;
  };
  float*    sh     = (float*)   alloc((size_t)N_EDGES * 9  * sizeof(float));
  _Float16* emb16  = (_Float16*)alloc((size_t)N_EDGES * 16 * sizeof(_Float16));
  _Float16* ea16   = (_Float16*)alloc((size_t)N_EDGES * 16 * sizeof(_Float16));
  float*    w      = (float*)   alloc((size_t)N_EDGES * 32 * sizeof(float));
  float*    agg    = (float*)   alloc((size_t)N_NODES * FEAT * sizeof(float));
  float*    xA     = (float*)   alloc((size_t)N_NODES * FEAT * sizeof(float));
  float*    xB     = (float*)   alloc((size_t)N_NODES * FEAT * sizeof(float));
  _Float16* W1p    = (_Float16*)alloc((size_t)LAYERS * 112 * 32 * sizeof(_Float16));
  _Float16* W2p    = (_Float16*)alloc((size_t)LAYERS * 32 * 128 * sizeof(_Float16));
  _Float16* Wattrp = (_Float16*)alloc((size_t)LAYERS * 32 * 32  * sizeof(_Float16));

  {
    int tot = LAYERS * (112*32 + 32*128 + 32*32);
    prep_weights_kernel<<<(tot + 255)/256, 256, 0, stream>>>(W1, W2, Wattr, W1p, W2p, Wattrp);
  }
  edge_features_kernel<<<(N_EDGES + 255)/256, 256, 0, stream>>>(edge_vec, edge_attr, sh, emb16, ea16);

  const float* xcur = node_input;
  float* bufs[2] = {xA, xB};
  for (int l = 0; l < LAYERS; ++l) {
    zero_kernel<<<((long)N_NODES*FEAT + 255)/256, 256, 0, stream>>>(agg, (long)N_NODES*FEAT);
    edge_weight_kernel<<<N_EDGES/(16*EW_WAVES), 256, 0, stream>>>(
        emb16, ea16, W1p + l*112*32, W2p + l*32*128, Wattrp + l*32*32, b1 + l*100, w);
    message_scatter_kernel<<<(int)(((long)N_EDGES*32)/256), 256, 0, stream>>>(
        xcur, sh, w, edge_src, edge_dst, agg);
    float* xn = bufs[l & 1];
    node_update_kernel<<<(N_NODES*SHD)/(16*8), 256, 0, stream>>>(
        xcur, agg, node_attr, Wself + l*CCH*CCH, Wout + l*CCH*CCH, xn);
    gate_kernel<<<(N_NODES*CCH + 255)/256, 256, 0, stream>>>(xn);
    xcur = xn;
  }
  zero_kernel<<<(out_size + 255)/256, 256, 0, stream>>>(out, (long)out_size);
  output_kernel<<<(int)(((long)N_NODES*FEAT + 255)/256), 256, 0, stream>>>(xcur, batch, out);
}
